// DeepseekV3MoE_86526411145672
// MI455X (gfx1250) — compile-verified, matrix operand order checked by reading, not run on hardware
//
#include <hip/hip_runtime.h>
#include <hip/hip_bf16.h>

// ---------------------------------------------------------------------------
// DeepseekV3 MoE for gfx1250 (MI455X), wave32 + WMMA bf16 + TDM gather.
//   T=2048 tokens, D=1024, E=16 experts, F=1024, K=4, NG=4, TKG=2, NSH=2.
// Router builds per-expert token lists; FFN kernel gathers token rows into
// LDS via the Tensor Data Mover (gather-mode descriptor, 16 row indices per
// op), then runs 32-row x 16-col WMMA bf16 tiles with software-pipelined
// weight streaming. Gate/up leading dimension is a template constant so all
// B-fragment loads fold into saddr + imm-offset addressing.
// ---------------------------------------------------------------------------

#define TOK  2048
#define DIM  1024
#define FDIM 1024
#define NEXP 16
#define NGRP 4
#define EPG  4
#define TOPK 4
#define NSH  2
#define RSCALE 2.5f
#define MT   32                         // token rows per block (2 WMMA row-tiles)

// Fixed LDS layout (identical on host/device passes):
//   [0,128K)    fp32 staging for TDM gather (MT x DIM)
//   [128K,192K) bf16 x tile (MT x DIM)
//   [192K,256K) bf16 act tile (MT x FDIM)
#define STAGE_BYTES (MT * DIM * 4)
#define XS_OFF      STAGE_BYTES
#define ACT_OFF     (STAGE_BYTES + MT * DIM * 2)
#define SMEM_BYTES  (STAGE_BYTES + MT * DIM * 2 + MT * FDIM * 2)   // 256 KB

typedef __attribute__((ext_vector_type(16))) __bf16 v16bf;
typedef __attribute__((ext_vector_type(8)))  __bf16 v8bf;
typedef __attribute__((ext_vector_type(8)))  float  v8f;
typedef __attribute__((ext_vector_type(4)))  unsigned int u32x4;
typedef __attribute__((ext_vector_type(8)))  int   i32x8;
typedef __attribute__((ext_vector_type(4)))  int   i32x4;

#if defined(__has_builtin)
#if __has_builtin(__builtin_amdgcn_tensor_load_to_lds) && \
    __has_builtin(__builtin_amdgcn_s_wait_tensorcnt)
#define HAVE_TDM 1
#endif
#endif
#ifndef HAVE_TDM
#define HAVE_TDM 0
#endif

// ---- WMMA helpers ----------------------------------------------------------

__device__ __forceinline__ v8f wmma_bf16(v16bf a, v16bf b, v8f c) {
  // D = A(16x32 bf16) * B(32x16 bf16) + C(16x16 f32)
  return __builtin_amdgcn_wmma_f32_16x16x32_bf16(
      false, a, false, b, (short)0, c, false, false);
}

// A fragment: 16x32 bf16 from LDS (row-major, compile-time stride LD).
// Lane m = lane&15 holds row M=m; K runs [8*half, +8) and [16+8*half, +8).
template <int LD>
__device__ __forceinline__ v16bf load_a_frag(const __bf16* base, int lane,
                                             int k0) {
  const int m = lane & 15, half = lane >> 4;
  const __bf16* p0 = base + m * LD + k0 + 8 * half;
  v16bf a;
#pragma unroll
  for (int j = 0; j < 8; ++j) {
    a[j]     = p0[j];
    a[8 + j] = p0[16 + j];
  }
  return a;
}

// B fragment: 32x16 from global fp32 weights W[k][n], compile-time row
// stride LDW, converted to bf16 in-register. 32-bit voffset + constant
// instruction offsets -> saddr-form global_load_b32.
template <int LDW>
__device__ __forceinline__ v16bf load_b_frag(const float* __restrict__ W,
                                             int lane, int k0, int n0) {
  const int n = lane & 15, half = lane >> 4;
  const unsigned off = (unsigned)((k0 + 8 * half) * LDW + n0 + n);
  // hint the chunk after next into cache (global_prefetch_b8)
  __builtin_prefetch(W + off + 64u * (unsigned)LDW, 0, 1);
  v16bf b;
#pragma unroll
  for (int j = 0; j < 8; ++j) {
    b[j]     = (__bf16)W[off + (unsigned)(j * LDW)];
    b[8 + j] = (__bf16)W[off + (unsigned)((16 + j) * LDW)];
  }
  return b;
}

__device__ __forceinline__ float fast_sigmoid(float x) {
  return __builtin_amdgcn_rcpf(1.0f + __expf(-x));
}

__device__ __forceinline__ v8bf cvt8_bf16(float4 lo, float4 hi) {
  v8bf o;
  o[0] = (__bf16)lo.x; o[1] = (__bf16)lo.y;
  o[2] = (__bf16)lo.z; o[3] = (__bf16)lo.w;
  o[4] = (__bf16)hi.x; o[5] = (__bf16)hi.y;
  o[6] = (__bf16)hi.z; o[7] = (__bf16)hi.w;
  return o;
}

#if HAVE_TDM
// Issue one TDM gather: 16 rows of x (fp32, DIM wide) -> LDS at lds_off.
// idxw = 8 dwords of packed 16-bit row indices.
__device__ __forceinline__ void tdm_gather16(const float* x, unsigned lds_off,
                                             const unsigned* idxw) {
  const unsigned long long ga = (unsigned long long)(uintptr_t)x;
  // D# group0: count=1, gather_mode=1 (16-bit indices), lds_addr, global_addr,
  // type=2 ("image") in bits 127:126.
  u32x4 g0 = { 1u | (1u << 31), lds_off,
               (unsigned)ga, (unsigned)(ga >> 32) | (2u << 30) };
  // D# group1: data_size=2 (4B) @bits17:16; tensor_dim0=1024 @79:48;
  // tensor_dim1=2048 @111:80; tile_dim0=1024 @127:112; tile_dim1=16 indices
  // @143:128; tensor_dim0_stride=1024 @207:160.
  i32x8 g1 = { 0x00020000, 0x04000000, 0x08000000, 0x04000000,
               16, 1024, 0, 0 };
  i32x4 g2 = { (int)idxw[0], (int)idxw[1], (int)idxw[2], (int)idxw[3] };
  i32x4 g3 = { (int)idxw[4], (int)idxw[5], (int)idxw[6], (int)idxw[7] };
#if defined(__clang_major__) && (__clang_major__ >= 23)
  i32x8 z = { 0, 0, 0, 0, 0, 0, 0, 0 };
  __builtin_amdgcn_tensor_load_to_lds(g0, g1, g2, g3, z, 0);
#else
  __builtin_amdgcn_tensor_load_to_lds(g0, g1, g2, g3, 0);
#endif
}
#endif

// ---- Kernel 1: zero output + per-expert counters ---------------------------

__global__ void moe_init_kernel(float* out, int n, int* counts) {
  int gid = blockIdx.x * blockDim.x + threadIdx.x;
  if (gid < NEXP) counts[gid] = 0;
  for (int i = gid; i < n; i += gridDim.x * blockDim.x) out[i] = 0.0f;
}

// ---- Kernel 2: router (sigmoid + group top-k) ------------------------------
// 256 threads = 16 tokens x 16 experts per block.

__global__ __launch_bounds__(256)
void moe_router_kernel(const float* __restrict__ x,
                       const float* __restrict__ rw,
                       const float* __restrict__ bias,
                       int* __restrict__ counts,
                       int* __restrict__ etok,
                       float* __restrict__ ewt) {
  __shared__ float sc[16][NEXP];
  const int tid = threadIdx.x;
  const int tl = tid >> 4;          // token within block
  const int e  = tid & 15;          // expert
  const int t  = blockIdx.x * 16 + tl;

  const float* xp = x + (size_t)t * DIM;
  const float* wp = rw + (size_t)e * DIM;
  float acc = 0.0f;
#pragma unroll 4
  for (int d = 0; d < DIM; d += 4) {
    float4 xv = *(const float4*)(xp + d);
    float4 wv = *(const float4*)(wp + d);
    acc += xv.x * wv.x + xv.y * wv.y + xv.z * wv.z + xv.w * wv.w;
  }
  sc[tl][e] = fast_sigmoid(acc);
  __syncthreads();

  if (e == 0) {
    float s[NEXP], sb[NEXP];
#pragma unroll
    for (int i = 0; i < NEXP; ++i) { s[i] = sc[tl][i]; sb[i] = s[i] + bias[i]; }

    // group score = sum of top-2 within each group of 4
    float gs[NGRP];
#pragma unroll
    for (int g = 0; g < NGRP; ++g) {
      float m1 = -1e30f, m2 = -1e30f;
#pragma unroll
      for (int j = 0; j < EPG; ++j) {
        float v = sb[g * EPG + j];
        if (v > m1) { m2 = m1; m1 = v; } else if (v > m2) { m2 = v; }
      }
      gs[g] = m1 + m2;
    }
    int g0 = 0;
#pragma unroll
    for (int g = 1; g < NGRP; ++g) if (gs[g] > gs[g0]) g0 = g;
    gs[g0] = -1e30f;
    int g1 = 0;
#pragma unroll
    for (int g = 1; g < NGRP; ++g) if (gs[g] > gs[g1]) g1 = g;

    float masked[NEXP];
#pragma unroll
    for (int i = 0; i < NEXP; ++i) {
      int g = i >> 2;
      masked[i] = (g == g0 || g == g1) ? sb[i] : -1.0f;
    }

    int   idx[TOPK];
    float tw[TOPK];
    float tsum = 1e-20f;
#pragma unroll
    for (int k = 0; k < TOPK; ++k) {
      int best = 0;
#pragma unroll
      for (int i = 1; i < NEXP; ++i) if (masked[i] > masked[best]) best = i;
      idx[k] = best;
      tw[k]  = s[best];            // raw sigmoid score
      tsum  += tw[k];
      masked[best] = -1e30f;
    }
    const float scale = RSCALE * __builtin_amdgcn_rcpf(tsum);
#pragma unroll
    for (int k = 0; k < TOPK; ++k) {
      int ee  = idx[k];
      int pos = atomicAdd(&counts[ee], 1);
      etok[ee * TOK + pos] = t;
      ewt [ee * TOK + pos] = tw[k] * scale;
    }
  }
}

// ---- Kernel 3: expert FFN (SwiGLU) with WMMA + TDM gather ------------------
// grid.x = 64 token tiles (32 rows each); grid.y = job (expert id, or shared
// half if SHARED). 256 threads = 8 wave32s; each wave owns 8 column tiles of
// 16 and both 16-row sub-tiles (B-fragment reuse). LDG = gate/up row stride.

template <int LDG, bool SHARED>
__global__ __launch_bounds__(256)
void moe_ffn_kernel(const float* __restrict__ x,
                    const float* __restrict__ gate_base,  // routed: [E,D,F]; shared: [D,NSH*F]
                    const float* __restrict__ up_base,
                    const float* __restrict__ down_base,  // routed: [E,F,D]; shared: [NSH*F,D]
                    const int*   __restrict__ counts,
                    const int*   __restrict__ etok,
                    const float* __restrict__ ewt,
                    float* __restrict__ out) {
  extern __shared__ __align__(16) char smem[];
  __bf16* xs   = (__bf16*)(smem + XS_OFF);    // MT x DIM  bf16 (64 KB)
  __bf16* acts = (__bf16*)(smem + ACT_OFF);   // MT x FDIM bf16 (64 KB)

  const int job  = blockIdx.y;
  const int tile = blockIdx.x;

  const float *G, *U, *Dn;
  int e = 0, cnt;
  if (!SHARED) {
    e   = job;
    cnt = counts[e];
    if (tile * MT >= cnt) return;     // uniform exit: no work for this tile
    G   = gate_base + (size_t)e * DIM * FDIM;
    U   = up_base   + (size_t)e * DIM * FDIM;
    Dn  = down_base + (size_t)e * FDIM * DIM;
  } else {
    cnt = TOK;                        // shared half: cols [job*F, +F)
    G   = gate_base + job * FDIM;     // column offset, row stride LDG=NSH*F
    U   = up_base   + job * FDIM;
    Dn  = down_base + (size_t)job * FDIM * DIM;
  }

  const int wave = threadIdx.x >> 5;
  const int lane = threadIdx.x & 31;
  const int half = lane >> 4;
  const int nlo  = lane & 15;

#if HAVE_TDM
  // ---- TDM gather: DMA MT scattered token rows (fp32) into LDS staging ----
  if (wave == 0) {
    unsigned idxw[16];
#pragma unroll
    for (int j = 0; j < 16; ++j) {
      const int r0 = 2 * j, r1 = 2 * j + 1;
      int t0, t1;
      if (SHARED) {
        t0 = tile * MT + r0;
        t1 = tile * MT + r1;
      } else {
        t0 = (tile * MT + r0 < cnt) ? etok[e * TOK + tile * MT + r0] : 0;
        t1 = (tile * MT + r1 < cnt) ? etok[e * TOK + tile * MT + r1] : 0;
      }
      idxw[j] = ((unsigned)t0 & 0xFFFFu) | (((unsigned)t1 & 0xFFFFu) << 16);
    }
    tdm_gather16(x, 0u, &idxw[0]);                          // rows 0..15
    tdm_gather16(x, (unsigned)(16 * DIM * 4), &idxw[8]);    // rows 16..31
    __builtin_amdgcn_s_wait_tensorcnt(0);
  }
  __syncthreads();
  // fp32 staging -> bf16 x tile, 8 elems/step (packed cvt + b128 store)
  {
    const float* stag = (const float*)smem;
    for (int i8 = threadIdx.x; i8 < MT * DIM / 8; i8 += blockDim.x) {
      const int i = i8 * 8;
      const int r = i >> 10;
      v8bf o;
      if (tile * MT + r < cnt) {
        o = cvt8_bf16(*(const float4*)(stag + i), *(const float4*)(stag + i + 4));
      } else {
#pragma unroll
        for (int j = 0; j < 8; ++j) o[j] = (__bf16)0.0f;
      }
      *(v8bf*)(xs + i) = o;
    }
  }
#else
  // ---- fallback: gather via per-lane global loads, 8 elems/step -----------
  for (int i8 = threadIdx.x; i8 < MT * DIM / 8; i8 += blockDim.x) {
    const int i = i8 * 8;
    const int r = i >> 10, c = i & (DIM - 1);
    const int tr = tile * MT + r;
    v8bf o;
    if (tr < cnt) {
      const int tok = SHARED ? tr : etok[e * TOK + tr];
      const float* xp = x + (size_t)tok * DIM + c;
      o = cvt8_bf16(*(const float4*)xp, *(const float4*)(xp + 4));
    } else {
#pragma unroll
      for (int j = 0; j < 8; ++j) o[j] = (__bf16)0.0f;
    }
    *(v8bf*)(xs + i) = o;
  }
#endif
  __syncthreads();

  // ---- gate/up projections + SwiGLU --------------------------------------
  for (int nt = wave; nt < FDIM / 16; nt += 8) {
    const int n0 = nt * 16;
    v8f hc[2] = {{}, {}};
    v8f uc[2] = {{}, {}};
    // software pipeline, 2x unrolled K (no register rotation copies)
    v16bf bg0 = load_b_frag<LDG>(G, lane, 0, n0);
    v16bf bu0 = load_b_frag<LDG>(U, lane, 0, n0);
    for (int k0 = 0; k0 < DIM; k0 += 64) {
      v16bf bg1 = load_b_frag<LDG>(G, lane, k0 + 32, n0);
      v16bf bu1 = load_b_frag<LDG>(U, lane, k0 + 32, n0);
#pragma unroll
      for (int rt = 0; rt < 2; ++rt) {
        v16bf a = load_a_frag<DIM>(xs + rt * 16 * DIM, lane, k0);
        hc[rt] = wmma_bf16(a, bg0, hc[rt]);
        uc[rt] = wmma_bf16(a, bu0, uc[rt]);
      }
      if (k0 + 64 < DIM) {
        bg0 = load_b_frag<LDG>(G, lane, k0 + 64, n0);
        bu0 = load_b_frag<LDG>(U, lane, k0 + 64, n0);
      }
#pragma unroll
      for (int rt = 0; rt < 2; ++rt) {
        v16bf a = load_a_frag<DIM>(xs + rt * 16 * DIM, lane, k0 + 32);
        hc[rt] = wmma_bf16(a, bg1, hc[rt]);
        uc[rt] = wmma_bf16(a, bu1, uc[rt]);
      }
    }
#pragma unroll
    for (int rt = 0; rt < 2; ++rt) {
#pragma unroll
      for (int v = 0; v < 8; ++v) {
        const int M = rt * 16 + v + 8 * half;   // C/D layout: row = vgpr+8*half
        const float h = hc[rt][v];
        const float sil = h * __builtin_amdgcn_rcpf(1.0f + __expf(-h));
        acts[M * FDIM + n0 + nlo] = (__bf16)(sil * uc[rt][v]);
      }
    }
  }
  __syncthreads();

  // ---- down projection + weighted scatter-add ----------------------------
  for (int dt = wave; dt < DIM / 16; dt += 8) {
    const int d0 = dt * 16;
    v8f yc[2] = {{}, {}};
    v16bf bd0 = load_b_frag<DIM>(Dn, lane, 0, d0);
    for (int k0 = 0; k0 < FDIM; k0 += 64) {
      v16bf bd1 = load_b_frag<DIM>(Dn, lane, k0 + 32, d0);
#pragma unroll
      for (int rt = 0; rt < 2; ++rt) {
        v16bf a = load_a_frag<FDIM>(acts + rt * 16 * FDIM, lane, k0);
        yc[rt] = wmma_bf16(a, bd0, yc[rt]);
      }
      if (k0 + 64 < FDIM) bd0 = load_b_frag<DIM>(Dn, lane, k0 + 64, d0);
#pragma unroll
      for (int rt = 0; rt < 2; ++rt) {
        v16bf a = load_a_frag<FDIM>(acts + rt * 16 * FDIM, lane, k0 + 32);
        yc[rt] = wmma_bf16(a, bd1, yc[rt]);
      }
    }
#pragma unroll
    for (int rt = 0; rt < 2; ++rt) {
#pragma unroll
      for (int v = 0; v < 8; ++v) {
        const int M = rt * 16 + v + 8 * half;
        const int tr = tile * MT + M;
        float w = 0.0f;
        int tok = 0;
        if (tr < cnt) {
          if (SHARED) { tok = tr; w = 1.0f; }
          else        { tok = etok[e * TOK + tr]; w = ewt[e * TOK + tr]; }
        }
        unsafeAtomicAdd(&out[(size_t)tok * DIM + d0 + nlo], yc[rt][v] * w);
      }
    }
  }
}

// ---------------------------------------------------------------------------

extern "C" void kernel_launch(void* const* d_in, const int* in_sizes, int n_in,
                              void* d_out, int out_size, void* d_ws, size_t ws_size,
                              hipStream_t stream) {
  const float* x        = (const float*)d_in[0];  // [1, 2048, 1024]
  const float* router_w = (const float*)d_in[1];  // [16, 1024]
  const float* bias     = (const float*)d_in[2];  // [16]
  const float* gate_w   = (const float*)d_in[3];  // [16, 1024, 1024]
  const float* up_w     = (const float*)d_in[4];  // [16, 1024, 1024]
  const float* down_w   = (const float*)d_in[5];  // [16, 1024, 1024]
  const float* sh_gate  = (const float*)d_in[6];  // [1024, 2048]
  const float* sh_up    = (const float*)d_in[7];  // [1024, 2048]
  const float* sh_down  = (const float*)d_in[8];  // [2048, 1024]
  float* out = (float*)d_out;                     // [1, 2048, 1024]

  // workspace layout
  char* ws = (char*)d_ws;
  int*   counts = (int*)ws;                               // 16 ints
  int*   etok   = (int*)(ws + 4096);                      // [E, T] ints
  float* ewt    = (float*)(ws + 4096 + NEXP * TOK * 4);   // [E, T] floats

  // 1) zero output + counters
  moe_init_kernel<<<2048, 256, 0, stream>>>(out, out_size, counts);

  // 2) routing + sparse dispatch lists
  moe_router_kernel<<<TOK / 16, 256, 0, stream>>>(x, router_w, bias,
                                                  counts, etok, ewt);

  // 3) expert FFNs: 16 routed experts (LDG=F), then 2 shared halves
  //    (LDG=NSH*F). 256 KB dynamic LDS each.
  moe_ffn_kernel<FDIM, false><<<dim3(TOK / MT, NEXP), 256, SMEM_BYTES, stream>>>(
      x, gate_w, up_w, down_w, counts, etok, ewt, out);
  moe_ffn_kernel<NSH * FDIM, true><<<dim3(TOK / MT, NSH), 256, SMEM_BYTES, stream>>>(
      x, sh_gate, sh_up, sh_down, counts, etok, ewt, out);
}
